// BlockSelfAttention_83674552860990
// MI455X (gfx1250) — matrix operand order repaired and checked by reading, not compiled, not run on hardware
//
#include <hip/hip_runtime.h>

// ---------------------------------------------------------------------------
// Fused chunked self-attention for MI455X (gfx1250, wave32, WMMA f16->f32).
// One workgroup (256 thr = 8 waves) per (batch, chunk, head) tile:
//   X_tile [256 x 64] -> scores = X X^T / 8 + mask -> softmax -> ctx = P X
// ~17 GFLOP vs 128 MB HBM traffic => marginally memory bound. X is read once,
// converted to f16 in LDS in BOTH row-major (Q/K fragments) and transposed
// (V B-fragments) layouts so every WMMA operand is a contiguous aligned
// 16B/32B ds load. Softmax in fp32 registers via 16-lane shfl_xor reductions.
// ---------------------------------------------------------------------------

typedef __attribute__((ext_vector_type(16))) _Float16 v16h;
typedef __attribute__((ext_vector_type(8)))  _Float16 v8h;
typedef __attribute__((ext_vector_type(4)))  _Float16 v4h;
typedef __attribute__((ext_vector_type(8)))  float    v8f;

#define Bsz   4
#define Dmod  1024
#define Hh    16
#define DH    64
#define CHUNK 256
#define NC    16

#define XS_STRIDE 80    // halves: 160B row stride (multiple of 32 -> v16h ok)
#define XT_STRIDE 272   // halves: 544B row stride (multiple of 32 -> v16h ok)
#define PW_STRIDE 48    // halves:  96B row stride (multiple of 16 -> v8h ok)

#define XS_OFF 0                                  // 256*80*2  = 40960 B
#define XT_OFF 40960                              //  64*272*2 = 34816 B
#define PW_OFF 75776                              // 8*16*48*2 = 12288 B
#define SMEM_BYTES 88064

__global__ __launch_bounds__(256, 1)
void block_attn_kernel(const float* __restrict__ X,
                       const float* __restrict__ mask,
                       float* __restrict__ out)
{
    extern __shared__ __align__(128) unsigned char smem_raw[];
    _Float16 (*Xs)[XS_STRIDE]     = (_Float16(*)[XS_STRIDE])(smem_raw + XS_OFF);
    _Float16 (*Xt)[XT_STRIDE]     = (_Float16(*)[XT_STRIDE])(smem_raw + XT_OFF);
    _Float16 (*Pw)[16][PW_STRIDE] = (_Float16(*)[16][PW_STRIDE])(smem_raw + PW_OFF);

    const int t    = threadIdx.x;
    const int lane = t & 31;
    const int w    = t >> 5;          // wave id 0..7
    const int half = lane >> 4;       // 0: lanes 0-15, 1: lanes 16-31
    const int l16  = lane & 15;

    const int blk      = blockIdx.x;          // ((b*NC + c)*H + h)
    const int h        = blk % Hh;
    const int bc       = blk / Hh;            // b*NC + c
    const long rowBase = (long)bc * CHUNK;    // == b*S + c*CHUNK  (S = NC*CHUNK)
    const int  featBase = h * DH;

    // ---- global -> LDS: 256x64 fp32 tile -> f16, row-major + transposed ----
    // 4096 float4 total, 16 per thread, coalesced global reads.
    for (int i = 0; i < 16; ++i) {
        const int f   = i * 256 + t;          // 0..4095
        const int row = f >> 4;
        const int c4  = (f & 15) << 2;        // 0,4,...,60
        const float4 v = *(const float4*)(X + (rowBase + row) * Dmod + featBase + c4);
        v4h hv;
        hv[0] = (_Float16)v.x; hv[1] = (_Float16)v.y;
        hv[2] = (_Float16)v.z; hv[3] = (_Float16)v.w;
        *(v4h*)&Xs[row][c4] = hv;             // one ds_store_b64
        Xt[c4 + 0][row] = hv[0];
        Xt[c4 + 1][row] = hv[1];
        Xt[c4 + 2][row] = hv[2];
        Xt[c4 + 3][row] = hv[3];
    }
    __syncthreads();

    // additive mask value per column tile (mask flat index = b*S + c*CHUNK + col)
    float mk[16];
    {
        const float* mrow = mask + rowBase;
        #pragma unroll
        for (int n = 0; n < 16; ++n) mk[n] = mrow[n * 16 + l16];
    }

    for (int pass = 0; pass < 2; ++pass) {
        const int qrow0 = w * 32 + pass * 16;   // this wave's 16 query rows

        // ---- A-fragments of Q: 16x32 slices over d (contiguous k = 16*half + s) ----
        const int m = qrow0 + l16;
        v16h aq[2];
        #pragma unroll
        for (int s2 = 0; s2 < 2; ++s2) {
            const int dbase = s2 * 32 + (half ? 16 : 0);
            const v8h lo = *(const v8h*)&Xs[m][dbase];
            const v8h hi = *(const v8h*)&Xs[m][dbase + 8];
            aq[s2] = __builtin_shufflevector(lo, hi,
                     0,1,2,3,4,5,6,7,8,9,10,11,12,13,14,15);
        }

        // ---- scores = Q K^T : 16 column tiles of 16, K-dim = 64 (2 wmma) ----
        v8f acc[16];
        #pragma unroll
        for (int n = 0; n < 16; ++n) {
            v8f c = {};
            #pragma unroll
            for (int s2 = 0; s2 < 2; ++s2) {
                // B-frag of K^T: 16 consecutive d-values of one key row (32B aligned)
                const int krow  = n * 16 + l16;
                const int dbase = s2 * 32 + (half ? 16 : 0);
                const v16h bk = *(const v16h*)&Xs[krow][dbase];
                c = __builtin_amdgcn_wmma_f32_16x16x32_f16(false, aq[s2], false, bk,
                                                           (short)0, c, false, false);
            }
            acc[n] = c;
        }

        // ---- softmax over 256 cols. C-layout: vgpr r <-> row r+8*half,
        //      lane%16 <-> col; shfl_xor(1|2|4|8) reduces within a row. ----
        const float scale = 0.125f;  // 1/sqrt(64)
        float rmax[8], rsum[8];
        #pragma unroll
        for (int r = 0; r < 8; ++r) rmax[r] = -3.4e38f;
        #pragma unroll
        for (int n = 0; n < 16; ++n)
            #pragma unroll
            for (int r = 0; r < 8; ++r) {
                const float s = acc[n][r] * scale + mk[n];
                acc[n][r] = s;
                rmax[r] = fmaxf(rmax[r], s);
            }
        #pragma unroll
        for (int r = 0; r < 8; ++r) {
            float v = rmax[r];
            v = fmaxf(v, __shfl_xor(v, 1));
            v = fmaxf(v, __shfl_xor(v, 2));
            v = fmaxf(v, __shfl_xor(v, 4));
            v = fmaxf(v, __shfl_xor(v, 8));
            rmax[r] = v;
            rsum[r] = 0.f;
        }
        #pragma unroll
        for (int n = 0; n < 16; ++n)
            #pragma unroll
            for (int r = 0; r < 8; ++r) {
                const float p = __expf(acc[n][r] - rmax[r]);
                acc[n][r] = p;   // acc now holds un-normalized probabilities
                rsum[r] += p;
            }
        #pragma unroll
        for (int r = 0; r < 8; ++r) {
            float v = rsum[r];
            v += __shfl_xor(v, 1);
            v += __shfl_xor(v, 2);
            v += __shfl_xor(v, 4);
            v += __shfl_xor(v, 8);
            rsum[r] = v;
        }

        // ---- ctx = P V : stream 32-wide key slabs of P through per-wave LDS
        //      (DS ops are in-order within a wave -> no barrier needed) ----
        v8f ctx[4];
        #pragma unroll
        for (int n = 0; n < 4; ++n) ctx[n] = (v8f){};

        #pragma unroll
        for (int kb = 0; kb < 8; ++kb) {
            // store P columns [kb*32, kb*32+32) as f16 (C-layout -> row-major)
            #pragma unroll
            for (int tt = 0; tt < 2; ++tt) {
                const int n = kb * 2 + tt;
                #pragma unroll
                for (int r = 0; r < 8; ++r)
                    Pw[w][r + 8 * half][tt * 16 + l16] = (_Float16)acc[n][r];
            }
            asm volatile("" ::: "memory");   // keep store->load program order

            // A-fragment of P (16 rows x 32 keys)
            v16h ap;
            {
                const int dbase = half ? 16 : 0;
                const v8h lo = *(const v8h*)&Pw[w][l16][dbase];
                const v8h hi = *(const v8h*)&Pw[w][l16][dbase + 8];
                ap = __builtin_shufflevector(lo, hi,
                     0,1,2,3,4,5,6,7,8,9,10,11,12,13,14,15);
            }
            // B-fragments of V from transposed tile: 16 consecutive keys of one
            // feature row, 32B aligned -> single v16h load (2x ds_load_b128)
            #pragma unroll
            for (int n = 0; n < 4; ++n) {
                const int key0 = kb * 32 + (half ? 16 : 0);
                const int feat = n * 16 + l16;
                const v16h bv = *(const v16h*)&Xt[feat][key0];
                ctx[n] = __builtin_amdgcn_wmma_f32_16x16x32_f16(false, ap, false, bv,
                                                                (short)0, ctx[n], false, false);
            }
        }

        // ---- normalize by row sum and write out in [B,S,D] layout ----
        #pragma unroll
        for (int r = 0; r < 8; ++r) rsum[r] = 1.0f / rsum[r];
        #pragma unroll
        for (int n = 0; n < 4; ++n)
            #pragma unroll
            for (int r = 0; r < 8; ++r) {
                const int row  = qrow0 + r + 8 * half;
                const int feat = featBase + n * 16 + l16;
                out[(rowBase + row) * Dmod + feat] = ctx[n][r] * rsum[r];
            }
    }
}

extern "C" void kernel_launch(void* const* d_in, const int* in_sizes, int n_in,
                              void* d_out, int out_size, void* d_ws, size_t ws_size,
                              hipStream_t stream) {
    const float* hidden = (const float*)d_in[0];  // [B,S,D] fp32
    const float* mask   = (const float*)d_in[1];  // [B,1,1,S] fp32
    float* out          = (float*)d_out;          // [B,S,D] fp32

    const int grid = Bsz * NC * Hh;               // 1024 tiles
    block_attn_kernel<<<grid, 256, SMEM_BYTES, stream>>>(hidden, mask, out);
}
